// HeteroMLP_22789096472877
// MI455X (gfx1250) — matrix operand order repaired and checked by reading, not compiled, not run on hardware
//
#include <hip/hip_runtime.h>

#define NN 32768
#define TT 4

typedef __attribute__((ext_vector_type(16))) __bf16 v16bf;
typedef __attribute__((ext_vector_type(8)))  float  v8f;

__device__ __forceinline__ unsigned short f32_to_bf16(float f){
  unsigned u = __builtin_bit_cast(unsigned, f);
  u += 0x7FFFu + ((u >> 16) & 1u);          // round-to-nearest-even
  return (unsigned short)(u >> 16);
}

__global__ void k_cvt_bf16(const float* __restrict__ src, unsigned short* __restrict__ dst, int n){
  int i = blockIdx.x * blockDim.x + threadIdx.x;
  if (i < n) dst[i] = f32_to_bf16(src[i]);
}

__global__ void k_zero(unsigned int* __restrict__ p, int n){
  int i = blockIdx.x * blockDim.x + threadIdx.x;
  if (i < n) p[i] = 0u;
}

__global__ void k_count(const int* __restrict__ types, int* __restrict__ counts, int n){
  __shared__ int lc[TT];
  if (threadIdx.x < TT) lc[threadIdx.x] = 0;
  __syncthreads();
  for (int i = blockIdx.x * blockDim.x + threadIdx.x; i < n; i += gridDim.x * blockDim.x)
    atomicAdd(&lc[types[i]], 1);
  __syncthreads();
  if (threadIdx.x < TT) atomicAdd(&counts[threadIdx.x], lc[threadIdx.x]);
}

// Per-type channel statistics: sums[t][c] += x, ssum[t][c] += x*x
__global__ void k_stats(const float* __restrict__ H, const int* __restrict__ types,
                        float* __restrict__ sums, float* __restrict__ ssum, int C){
  const int col  = blockIdx.x * 64 + (threadIdx.x & 63);
  const int sub  = threadIdx.x >> 6;           // 0..3
  const int row0 = blockIdx.y * 1024;
  float s[TT] = {0.f,0.f,0.f,0.f}, q[TT] = {0.f,0.f,0.f,0.f};
  for (int r = sub; r < 1024; r += 4){
    const int row = row0 + r;
    const float v = H[(size_t)row * C + col];
    const int tt = types[row];
#pragma unroll
    for (int t = 0; t < TT; ++t){
      const float m = (tt == t) ? 1.f : 0.f;   // branch-free: v_cndmask + fma
      s[t] += m * v;
      q[t] += m * v * v;
    }
  }
#pragma unroll
  for (int t = 0; t < TT; ++t){
    atomicAdd(&sums[t * C + col], s[t]);
    atomicAdd(&ssum[t * C + col], q[t]);
  }
}

// Type-masked WMMA GEMM with optional fused (per-type BN + LeakyReLU) on the INPUT.
// Out[n, :] = act(norm(X[n,:])) @ W[type[n]] + bias[type[n]]
// 256 threads = 8 waves; block tile 256 rows x 64 cols; each wave owns TWO 16-row
// strips (rOff and rOff+128) so every B fragment feeds two WMMAs. K step = 32.
__launch_bounds__(256)
__global__ void k_hetero_gemm(const float* __restrict__ X,             // [M, K] fp32
                              const unsigned short* __restrict__ Wb,   // [T, K, Ncols] bf16
                              const float* __restrict__ bias,          // [T, Ncols]
                              const int*  __restrict__ types,          // [M]
                              const float* __restrict__ sums,          // [T, K] (fused BN) or null
                              const float* __restrict__ ssum,          // [T, K]
                              const int*  __restrict__ counts,         // [T]
                              const float* __restrict__ gamma,         // [T, K]
                              const float* __restrict__ beta,          // [T, K]
                              int doNorm,
                              float* __restrict__ Out,                 // [M, Ncols]
                              int K, int Ncols){
  __shared__ unsigned short As[256 * 32];       // 16 KB : A tile, bf16, k-contiguous per row
  __shared__ unsigned short Bs[TT * 64 * 32];   // 16 KB : B tiles per type, k-contiguous per col

  const int tid  = threadIdx.x;
  const int lane = tid & 31;
  const int wave = tid >> 5;                    // 0..7  -> strips (wave*16) and (wave*16+128)
  const int rowBase = blockIdx.x * 256;
  const int colBase = blockIdx.y * 64;

  v8f acc[2][4];
#pragma unroll
  for (int s = 0; s < 2; ++s)
#pragma unroll
    for (int c = 0; c < 4; ++c)
#pragma unroll
      for (int r = 0; r < 8; ++r) acc[s][c][r] = 0.f;

  const int rOff    = wave * 16;
  const int myType0 = types[rowBase + rOff +       (lane & 15)];  // A-frag row = lane&15
  const int myType1 = types[rowBase + rOff + 128 + (lane & 15)];

  const int aRow0 = tid >> 2;          // 0..63 (four passes cover 256 rows)
  const int aK    = (tid & 3) * 8;     // 8 contiguous k per thread

  for (int k0 = 0; k0 < K; k0 += 32){
    // ---- stage A: fp32 -> (fused BN + LeakyReLU) -> bf16  (256 x 32)
#pragma unroll
    for (int it = 0; it < 4; ++it){
      const int row  = aRow0 + it * 64;
      const int grow = rowBase + row;
      const int t    = types[grow];
      const float* xp = &X[(size_t)grow * K + k0 + aK];
      float v[8];
#pragma unroll
      for (int e = 0; e < 8; ++e) v[e] = xp[e];
      if (doNorm){
        float cnt = (float)counts[t]; if (cnt < 1.f) cnt = 1.f;
#pragma unroll
        for (int e = 0; e < 8; ++e){
          const int ch = k0 + aK + e;
          const float mean = sums[t * K + ch] / cnt;
          const float var  = ssum[t * K + ch] / cnt - mean * mean;
          const float xn   = (v[e] - mean) * rsqrtf(var + 1e-5f);
          const float y    = xn * gamma[t * K + ch] + beta[t * K + ch];
          v[e] = (y > 0.f) ? y : 0.01f * y;       // LeakyReLU
        }
      }
      unsigned short* ap = &As[row * 32 + aK];
#pragma unroll
      for (int e = 0; e < 8; ++e) ap[e] = f32_to_bf16(v[e]);
    }
    // ---- stage B for all 4 types: global row-major (col-contig) -> LDS col-major (k-contig)
#pragma unroll
    for (int it = 0; it < 4; ++it){
      const int lin8 = (tid + 256 * it) * 8;      // 8 consecutive cols per thread
      const int col  = lin8 & 63;
      const int k    = (lin8 >> 6) & 31;
      const int t    = lin8 >> 11;
      const unsigned short* wp = &Wb[(size_t)(t * K + k0 + k) * Ncols + colBase + col];
#pragma unroll
      for (int e = 0; e < 8; ++e)
        Bs[(t * 64 + col + e) * 32 + k] = wp[e];
    }
    __syncthreads();

    // ---- A fragments (16-bit A 16x32 layout: row = lane&15; K striped over halves)
    v16bf a0, a1;
    {
      const unsigned short* Ar0 = &As[(rOff +       (lane & 15)) * 32];
      const unsigned short* Ar1 = &As[(rOff + 128 + (lane & 15)) * 32];
      const int khalf = (lane >= 16) ? 8 : 0;
#pragma unroll
      for (int j = 0; j < 8; ++j){
        const int kk = ((j < 4) ? 0 : 16) + 2 * (j & 3) + khalf;
        a0[2 * j]     = __builtin_bit_cast(__bf16, Ar0[kk]);
        a0[2 * j + 1] = __builtin_bit_cast(__bf16, Ar0[kk + 1]);
        a1[2 * j]     = __builtin_bit_cast(__bf16, Ar1[kk]);
        a1[2 * j + 1] = __builtin_bit_cast(__bf16, Ar1[kk + 1]);
      }
    }
    v16bf zf;
#pragma unroll
    for (int e = 0; e < 16; ++e) zf[e] = (__bf16)0.0f;

    // ---- 4 type-masked passes; every B fragment feeds 2 WMMAs (both row strips)
#pragma unroll
    for (int t = 0; t < TT; ++t){
      const v16bf at0 = (myType0 == t) ? a0 : zf;
      const v16bf at1 = (myType1 == t) ? a1 : zf;
#pragma unroll
      for (int c = 0; c < 4; ++c){
        v16bf bfrag;   // B 32x16 layout: col = lane&15; K = (h<8?0:16)+(lane>=16?8:0)+(h&7)
        const unsigned short* Br =
            &Bs[(t * 64 + c * 16 + (lane & 15)) * 32 + ((lane >= 16) ? 8 : 0)];
#pragma unroll
        for (int h = 0; h < 16; ++h) bfrag[h] = __builtin_bit_cast(__bf16, Br[((h < 8) ? 0 : 16) + (h & 7)]);
        acc[0][c] = __builtin_amdgcn_wmma_f32_16x16x32_bf16(false, at0, false, bfrag,
                                                            (short)0, acc[0][c], false, false);
        acc[1][c] = __builtin_amdgcn_wmma_f32_16x16x32_bf16(false, at1, false, bfrag,
                                                            (short)0, acc[1][c], false, false);
      }
    }
    __syncthreads();
  }

  // ---- epilogue: per-row-type bias, store fp32
  // C layout: VGPR r -> M = r (lanes 0-15) / r+8 (lanes 16-31); N = lane&15
#pragma unroll
  for (int s = 0; s < 2; ++s){
    const int rowTop = rowBase + rOff + s * 128 + ((lane >= 16) ? 8 : 0);
    int rt[8];
#pragma unroll
    for (int r = 0; r < 8; ++r) rt[r] = types[rowTop + r];
#pragma unroll
    for (int c = 0; c < 4; ++c){
      const int col = colBase + c * 16 + (lane & 15);
#pragma unroll
      for (int r = 0; r < 8; ++r)
        Out[(size_t)(rowTop + r) * Ncols + col] = acc[s][c][r] + bias[rt[r] * Ncols + col];
    }
  }
}

extern "C" void kernel_launch(void* const* d_in, const int* in_sizes, int n_in,
                              void* d_out, int out_size, void* d_ws, size_t ws_size,
                              hipStream_t stream){
  (void)in_sizes; (void)n_in; (void)out_size; (void)ws_size;
  const float* x     = (const float*)d_in[0];
  const int*   types = (const int*)  d_in[1];
  const float* W1 = (const float*)d_in[2];
  const float* b1 = (const float*)d_in[3];
  const float* W2 = (const float*)d_in[4];
  const float* b2 = (const float*)d_in[5];
  const float* W3 = (const float*)d_in[6];
  const float* b3 = (const float*)d_in[7];
  const float* g1 = (const float*)d_in[8];
  const float* be1= (const float*)d_in[9];
  const float* g2 = (const float*)d_in[10];
  const float* be2= (const float*)d_in[11];
  float* out = (float*)d_out;

  const int N = NN, CIN = 256, H = 512, COUT = 256, T = TT;

  char* ws = (char*)d_ws; size_t off = 0;
  auto alloc = [&](size_t bytes) -> void* {
    void* p = ws + off; off += (bytes + 255) & ~(size_t)255; return p;
  };
  unsigned short* wb1 = (unsigned short*)alloc((size_t)T * CIN * H   * 2);
  unsigned short* wb2 = (unsigned short*)alloc((size_t)T * H   * H   * 2);
  unsigned short* wb3 = (unsigned short*)alloc((size_t)T * H   * COUT* 2);
  float* h1 = (float*)alloc((size_t)N * H * 4);
  float* h2 = (float*)alloc((size_t)N * H * 4);
  float* sums1 = (float*)alloc((size_t)T * H * 4);   // stats block is contiguous:
  float* ss1   = (float*)alloc((size_t)T * H * 4);   // 4 x 8192B + padded counts
  float* sums2 = (float*)alloc((size_t)T * H * 4);
  float* ss2   = (float*)alloc((size_t)T * H * 4);
  int*   counts= (int*)  alloc((size_t)T * 4);

  // bf16 weight conversion (done every call; cheap, deterministic)
  { int n = T*CIN*H;  k_cvt_bf16<<<(n+255)/256,256,0,stream>>>(W1, wb1, n); }
  { int n = T*H*H;    k_cvt_bf16<<<(n+255)/256,256,0,stream>>>(W2, wb2, n); }
  { int n = T*H*COUT; k_cvt_bf16<<<(n+255)/256,256,0,stream>>>(W3, wb3, n); }

  // zero stats + counts (contiguous region: 4*8192 B + 256 B pad = 8256 dwords)
  k_zero<<<(8256 + 255)/256, 256, 0, stream>>>((unsigned int*)sums1, 8256);
  k_count<<<64, 256, 0, stream>>>(types, counts, N);

  const dim3 blk(256);
  const dim3 gA(N/256, H/64);      // (128, 8)
  const dim3 gS(H/64, N/1024);     // (8, 32)
  const dim3 gC(N/256, COUT/64);   // (128, 4)

  // L1: x @ W1 + b1  (no input norm)
  k_hetero_gemm<<<gA, blk, 0, stream>>>(x, wb1, b1, types,
                                        nullptr, nullptr, nullptr, nullptr, nullptr, 0,
                                        h1, CIN, H);
  // per-type stats of h1
  k_stats<<<gS, blk, 0, stream>>>(h1, types, sums1, ss1, H);
  // L2: act(BN1(h1)) @ W2 + b2, BN+LeakyReLU fused into A-staging
  k_hetero_gemm<<<gA, blk, 0, stream>>>(h1, wb2, b2, types,
                                        sums1, ss1, counts, g1, be1, 1,
                                        h2, H, H);
  // per-type stats of h2
  k_stats<<<gS, blk, 0, stream>>>(h2, types, sums2, ss2, H);
  // L3: act(BN2(h2)) @ W3 + b3 -> out (plain last layer)
  k_hetero_gemm<<<gC, blk, 0, stream>>>(h2, wb3, b3, types,
                                        sums2, ss2, counts, g2, be2, 1,
                                        out, H, COUT);
}
